// Attention_5892695130441
// MI455X (gfx1250) — matrix operand order repaired and checked by reading, not compiled, not run on hardware
//
#include <hip/hip_runtime.h>

typedef unsigned short u16;
typedef unsigned int   u32;

typedef __bf16 bf16x16 __attribute__((ext_vector_type(16)));
typedef float  f32x8   __attribute__((ext_vector_type(8)));
typedef unsigned int u32x4 __attribute__((ext_vector_type(4)));
typedef int          i32x4 __attribute__((ext_vector_type(4)));
typedef int          i32x8 __attribute__((ext_vector_type(8)));

union Frag16 {
    uint4   q[2];   // 32 bytes = 8 VGPRs
    bf16x16 v;
};

__device__ __forceinline__ u16 f32_to_bf16(float f) {
    union { float f; u32 u; } c; c.f = f;
    u32 u = c.u;
    u32 r = (u + 0x7FFFu + ((u >> 16) & 1u)) >> 16;   // round-to-nearest-even
    return (u16)r;
}

__device__ __forceinline__ f32x8 wmma_bf16(const Frag16& a, const Frag16& b, f32x8 c) {
    return __builtin_amdgcn_wmma_f32_16x16x32_bf16(false, a.v, false, b.v,
                                                   (short)0, c, false, false);
}

// ---------------------------------------------------------------------------
// Tensor Data Mover: 2D tile (tile0 x tile1 elements of 2 bytes) from global
// (row stride stride0 elems) into LDS at lds_off, with LDS padding
// (pad codes per ISA 8.4: interval 2^(padi+1) DWORDs, amount pada+1 DWORDs),
// and hardware zero-fill for reads beyond (td0, td1).
// ---------------------------------------------------------------------------
__device__ __forceinline__ void tdm_load_2d(u32 lds_off, const u16* gp,
                                            u32 td0, u32 td1,
                                            u32 tile0, u32 tile1,
                                            u32 stride0,
                                            u32 padi, u32 pada, bool pad_en) {
    unsigned long long ga = (unsigned long long)(uintptr_t)gp;
    u32x4 g0;
    g0.x = 1u;                                          // count=1, user mode
    g0.y = lds_off;                                     // lds_addr
    g0.z = (u32)ga;                                     // global_addr[31:0]
    g0.w = (u32)((ga >> 32) & 0x01FFFFFFu) | (2u << 30); // addr[56:32] | type=2
    i32x8 g1;
    g1[0] = (int)((1u << 16) |                          // data_size = 2 bytes
                  ((pad_en ? 1u : 0u) << 20) | (padi << 22) | (pada << 25));
    g1[1] = (int)(td0 << 16);                           // tensor_dim0[15:0]
    g1[2] = (int)((td0 >> 16) | (td1 << 16));           // dim0 hi | dim1 lo
    g1[3] = (int)((td1 >> 16) | (tile0 << 16));         // dim1 hi | tile_dim0
    g1[4] = (int)(tile1 & 0xFFFFu);                     // tile_dim1, tile_dim2=0
    g1[5] = (int)stride0;                               // tensor_dim0_stride lo
    g1[6] = 0;
    g1[7] = 0;
    i32x4 gz = {0, 0, 0, 0};
#if defined(__clang_major__) && (__clang_major__ >= 23)
    i32x8 gz8 = {0, 0, 0, 0, 0, 0, 0, 0};
    __builtin_amdgcn_tensor_load_to_lds(g0, g1, gz, gz, gz8, 0);
#else
    __builtin_amdgcn_tensor_load_to_lds(g0, g1, gz, gz, 0);
#endif
}

// ---------------------------------------------------------------------------
__global__ void cvt_kernel(const float* __restrict__ in, u16* __restrict__ out, int n) {
    int i = blockIdx.x * 256 + threadIdx.x;
    if (i < n) out[i] = f32_to_bf16(in[i]);
}

// trans[h][q][k] = sum_w table[q,k,w]*params[w,h] + (q==k), bf16
__global__ void trans_kernel(const float* __restrict__ table,
                             const float* __restrict__ params,
                             u16* __restrict__ transb) {
    __shared__ float sp[72 * 12];
    for (int i = threadIdx.x; i < 72 * 12; i += 256) sp[i] = params[i];
    __syncthreads();
    int idx = blockIdx.x * 256 + threadIdx.x;       // q*196 + k
    if (idx >= 196 * 196) return;
    int q = idx / 196, k = idx % 196;
    const float* t = table + idx * 72;
    float acc[12];
#pragma unroll
    for (int h = 0; h < 12; ++h) acc[h] = 0.f;
    for (int w = 0; w < 72; ++w) {
        float tv = t[w];
#pragma unroll
        for (int h = 0; h < 12; ++h) acc[h] += tv * sp[w * 12 + h];
    }
    float eye = (q == k) ? 1.f : 0.f;
#pragma unroll
    for (int h = 0; h < 12; ++h)
        transb[(h * 196 + q) * 196 + k] = f32_to_bf16(acc[h] + eye);
}

__global__ void copy_cls_kernel(const u16* __restrict__ qbuf, u16* __restrict__ qprime) {
    int i = blockIdx.x * 256 + threadIdx.x;
    if (i < 768 * 64) {
        int z = i >> 6, d = i & 63;
        qprime[z * 12608 + d] = qbuf[z * 12608 + d];
    }
}

// ---------------------------------------------------------------------------
// bf16 GEMM, 128x128 block tile, BK=32. A-tile staged by TDM (double-buffered,
// hardware edge zero-fill + LDS pad stride 40); B-tile transposed via VALU.
// MODE 0: QKV; MODE 1: proj (f32 out); MODE 2: q-transform (batched over z)
// ---------------------------------------------------------------------------
template <int MODE>
__global__ __launch_bounds__(256)
void gemm_bf16(const u16* __restrict__ A, const u16* __restrict__ B,
               const float* __restrict__ bias, void* __restrict__ Out) {
    constexpr int M   = (MODE == 2) ? 196 : 12608;
    constexpr int N   = (MODE == 0) ? 2304 : (MODE == 1 ? 768 : 64);
    constexpr int K   = (MODE == 2) ? 196 : 768;
    constexpr int LDA = (MODE == 2) ? 196 : 768;
    constexpr int LDB = (MODE == 0) ? 2304 : (MODE == 1 ? 768 : 64);
    constexpr int AST = 40;    // LDS row stride (bf16): 32 + 8 pad (TDM pad 3/3)
    constexpr int BST = 40;

    __shared__ u16 Asm[2 * 128 * AST];
    __shared__ u16 Bsm[2 * 128 * BST];

    const int z = blockIdx.z;
    const u16* Ab = A;
    const u16* Bb = B;
    if (MODE == 2) { Ab = A + (z % 12) * (196 * 196); Bb = B + z * 12608 + 64; }

    const int m0    = blockIdx.y * 128;
    const int n0    = blockIdx.x * 128;
    const int tid   = threadIdx.x;
    const int lane  = tid & 31;
    const int wave  = tid >> 5;
    const int waveM = wave >> 2;
    const int waveN = wave & 3;
    const int halfl = lane >> 4;
    const int l16   = lane & 15;
    const int nK    = (K + 31) / 32;

    const f32x8 vzero = {0.f, 0.f, 0.f, 0.f, 0.f, 0.f, 0.f, 0.f};
    f32x8 acc[4][2];
#pragma unroll
    for (int i = 0; i < 4; ++i)
#pragma unroll
        for (int j = 0; j < 2; ++j) acc[i][j] = vzero;

    u32 breg[8];
    auto loadB = [&](int k0) {
#pragma unroll
        for (int i = 0; i < 8; ++i) {
            int lin = tid + i * 256;
            int r = lin >> 6, cp = (lin & 63) * 2;
            int gk = k0 + r, gn = n0 + cp;
            breg[i] = (gk < K && gn < N) ? *(const u32*)(Bb + (size_t)gk * LDB + gn) : 0u;
        }
    };
    auto storeB = [&](int buf) {
        u16* bp = Bsm + buf * (128 * BST);
#pragma unroll
        for (int i = 0; i < 8; ++i) {
            int lin = tid + i * 256;
            int r = lin >> 6, cp = (lin & 63) * 2;
            bp[cp * BST + r]       = (u16)(breg[i] & 0xFFFFu);
            bp[(cp + 1) * BST + r] = (u16)(breg[i] >> 16);
        }
    };
    auto issueA = [&](int k0, int buf) {
        tdm_load_2d((u32)(uintptr_t)(Asm + buf * (128 * AST)),
                    Ab + (size_t)m0 * LDA + k0,
                    (u32)(K - k0), (u32)(M - m0),   // hw zero-fill past edges
                    32u, 128u, (u32)LDA,
                    3u, 3u, true);                  // pad: every 16 DW, +4 DW
    };

    // prologue: stage tile 0
    if (wave == 0) issueA(0, 0);
    loadB(0);
    storeB(0);
    if (wave == 0) __builtin_amdgcn_s_wait_tensorcnt(0);
    __syncthreads();

    for (int kk = 0; kk < nK; ++kk) {
        const int buf = kk & 1;
        const bool more = (kk + 1) < nK;
        if (more) {
            if (wave == 0) issueA((kk + 1) * 32, buf ^ 1);   // overlap with WMMA
            loadB((kk + 1) * 32);
        }
        const u16* As = Asm + buf * (128 * AST);
        const u16* Bs = Bsm + buf * (128 * BST);

        Frag16 bfr[2];
#pragma unroll
        for (int nt = 0; nt < 2; ++nt) {
            const u16* p = Bs + (waveN * 32 + nt * 16 + l16) * BST + halfl * 8;
            bfr[nt].q[0] = *(const uint4*)p;
            bfr[nt].q[1] = *(const uint4*)(p + 16);
        }
#pragma unroll
        for (int mt = 0; mt < 4; ++mt) {
            Frag16 af;
            const u16* p = As + (waveM * 64 + mt * 16 + l16) * AST + halfl * 8;
            af.q[0] = *(const uint4*)p;
            af.q[1] = *(const uint4*)(p + 16);
#pragma unroll
            for (int nt = 0; nt < 2; ++nt)
                acc[mt][nt] = wmma_bf16(af, bfr[nt], acc[mt][nt]);
        }
        if (more) storeB(buf ^ 1);
        if (wave == 0) __builtin_amdgcn_s_wait_tensorcnt(0);
        __syncthreads();
    }

    // epilogue: C/D layout -> m = r + 8*halfl, n = l16
#pragma unroll
    for (int mt = 0; mt < 4; ++mt) {
#pragma unroll
        for (int nt = 0; nt < 2; ++nt) {
#pragma unroll
            for (int r = 0; r < 8; ++r) {
                int gm = m0 + waveM * 64 + mt * 16 + halfl * 8 + r;
                int gn = n0 + waveN * 32 + nt * 16 + l16;
                float v = acc[mt][nt][r];
                if constexpr (MODE == 0) {
                    if (gm < M) {
                        v += bias[gn];
                        int which = gn / 768, rem = gn % 768;
                        int hh = rem >> 6, d = rem & 63;
                        int bb = gm / 197, nn = gm % 197;
                        ((u16*)Out)[((which * 768 + bb * 12 + hh) * 197 + nn) * 64 + d] =
                            f32_to_bf16(v);
                    }
                } else if constexpr (MODE == 1) {
                    if (gm < M) ((float*)Out)[gm * 768 + gn] = v + bias[gn];
                } else {
                    if (gm < 196 && gn < 64)
                        ((u16*)Out)[z * 12608 + (1 + gm) * 64 + gn] = f32_to_bf16(v);
                }
            }
        }
    }
}

// ---------------------------------------------------------------------------
// Fused attention: one block (4 waves) per (b,h). K staged by TDM (pad 4/3 ->
// stride 72); V transposed manually; each wave owns 16 q rows.
// ---------------------------------------------------------------------------
#define KST 72
#define VST 224
#define SST 212
#define PST 224
#define ATTN_SMEM (208*KST*2 + 64*VST*2 + 4*16*SST*4 + 4*16*PST*2)

__global__ __launch_bounds__(128)
void attn_kernel(const u16* __restrict__ qprime, const u16* __restrict__ qkvbuf,
                 u16* __restrict__ outb) {
    extern __shared__ char smem[];
    u16*   Ksm = (u16*)smem;                       // [208][KST]
    u16*   Vt  = Ksm + 208 * KST;                  // [64][VST]
    float* Ssm = (float*)(Vt + 64 * VST);          // 4 x [16][SST]
    u16*   Psm = (u16*)(Ssm + 4 * 16 * SST);       // 4 x [16][PST]

    const int z = blockIdx.x;                      // b*12 + h
    const int b = z / 12, h = z % 12;
    const int tid = threadIdx.x, lane = tid & 31, wave = tid >> 5;
    const int halfl = lane >> 4, l16 = lane & 15;

    const u16* Kg = qkvbuf + (768 + z) * 12608;
    const u16* Vg = qkvbuf + (2 * 768 + z) * 12608;
    const u16* Qz = qprime + z * 12608;

    for (int i = tid; i < 208 * KST / 2; i += 128) ((u32*)Ksm)[i] = 0;
    for (int i = tid; i < 64 * VST / 2; i += 128) ((u32*)Vt)[i] = 0;
    __syncthreads();
    if (wave == 0)   // K[197][64] -> LDS stride 72, via Tensor Data Mover
        tdm_load_2d((u32)(uintptr_t)Ksm, Kg, 64u, 197u, 64u, 197u, 64u, 4u, 3u, true);
    for (int i = tid; i < 197 * 32; i += 128) {    // V transposed (VALU path)
        int n = i >> 5, dp = (i & 31) * 2;
        u32 val = *(const u32*)(Vg + n * 64 + dp);
        Vt[dp * VST + n]       = (u16)(val & 0xFFFFu);
        Vt[(dp + 1) * VST + n] = (u16)(val >> 16);
    }
    if (wave == 0) __builtin_amdgcn_s_wait_tensorcnt(0);
    __syncthreads();

    float* Sw = Ssm + wave * 16 * SST;
    u16*   Pw = Psm + wave * 16 * PST;
    const f32x8 vzero = {0.f, 0.f, 0.f, 0.f, 0.f, 0.f, 0.f, 0.f};

    for (int t = wave; t < 13; t += 4) {
        int qr = t * 16 + l16; if (qr > 196) qr = 196;
        Frag16 aq[2];
#pragma unroll
        for (int c = 0; c < 2; ++c) {
            const u16* p = Qz + qr * 64 + c * 32 + halfl * 8;
            aq[c].q[0] = *(const uint4*)p;
            aq[c].q[1] = *(const uint4*)(p + 16);
        }
        // --- S = scale * Q K^T ---
        for (int nt = 0; nt < 13; ++nt) {
            Frag16 bk0, bk1;   // both 32-K chunks loaded before the WMMA pair
            const u16* p = Ksm + (nt * 16 + l16) * KST + halfl * 8;
            bk0.q[0] = *(const uint4*)p;
            bk0.q[1] = *(const uint4*)(p + 16);
            bk1.q[0] = *(const uint4*)(p + 32);
            bk1.q[1] = *(const uint4*)(p + 48);
            f32x8 s = wmma_bf16(aq[0], bk0, vzero);
            s = wmma_bf16(aq[1], bk1, s);
            int n = nt * 16 + l16;
#pragma unroll
            for (int r = 0; r < 8; ++r) {
                int m = halfl * 8 + r;
                Sw[m * SST + n] = (n < 197) ? s[r] * 0.125f : -3.0e38f;
            }
        }
        asm volatile("s_wait_dscnt 0" ::: "memory");

        // --- softmax: 2 lanes per row ---
        {
            int row = lane >> 1, seg = lane & 1;
            const float* Sr = Sw + row * SST + seg * 104;
            float mx = -3.0e38f;
            for (int i = 0; i < 104; ++i) mx = fmaxf(mx, Sr[i]);
            mx = fmaxf(mx, __shfl_xor(mx, 1, 32));
            float sum = 0.f;
            for (int i = 0; i < 104; ++i) sum += __expf(Sr[i] - mx);
            sum += __shfl_xor(sum, 1, 32);
            float rinv = 1.f / sum;
            u16* Pr = Pw + row * PST + seg * 104;
            for (int i = 0; i < 104; i += 2) {
                u32 p0 = f32_to_bf16(__expf(Sr[i] - mx) * rinv);
                u32 p1 = f32_to_bf16(__expf(Sr[i + 1] - mx) * rinv);
                *(u32*)(Pr + i) = p0 | (p1 << 16);
            }
            if (seg == 1)
                for (int i = 104; i < 120; i += 2) *(u32*)(Pr + i) = 0;
        }
        asm volatile("s_wait_dscnt 0" ::: "memory");

        // --- O = P V ---
        f32x8 o[4];
#pragma unroll
        for (int dt = 0; dt < 4; ++dt) o[dt] = vzero;
        for (int kc = 0; kc < 7; ++kc) {
            int k0 = kc * 32;
            Frag16 ap;
            const u16* pp = Pw + l16 * PST + k0 + halfl * 8;
            ap.q[0] = *(const uint4*)pp;
            ap.q[1] = *(const uint4*)(pp + 16);
#pragma unroll
            for (int dt = 0; dt < 4; ++dt) {
                Frag16 bv;
                const u16* pv = Vt + (dt * 16 + l16) * VST + k0 + halfl * 8;
                bv.q[0] = *(const uint4*)pv;
                bv.q[1] = *(const uint4*)(pv + 16);
                o[dt] = wmma_bf16(ap, bv, o[dt]);
            }
        }
#pragma unroll
        for (int dt = 0; dt < 4; ++dt) {
#pragma unroll
            for (int r = 0; r < 8; ++r) {
                int m = t * 16 + halfl * 8 + r;
                if (m < 197) {
                    int d = dt * 16 + l16;
                    outb[(b * 197 + m) * 768 + h * 64 + d] = f32_to_bf16(o[dt][r]);
                }
            }
        }
    }
}

// ---------------------------------------------------------------------------
extern "C" void kernel_launch(void* const* d_in, const int* in_sizes, int n_in,
                              void* d_out, int out_size, void* d_ws, size_t ws_size,
                              hipStream_t stream) {
    const float* x     = (const float*)d_in[0];
    const float* qkvw  = (const float*)d_in[1];
    const float* qkvb  = (const float*)d_in[2];
    const float* projw = (const float*)d_in[3];
    const float* projb = (const float*)d_in[4];
    const float* stab  = (const float*)d_in[5];
    const float* spar  = (const float*)d_in[6];

    char* ws = (char*)d_ws;
    size_t off = 0;
    auto carve = [&](size_t bytes) -> char* {
        char* p = ws + off;
        off += (bytes + 255) & ~(size_t)255;
        return p;
    };
    u16* qkvbuf = (u16*)carve(3ull * 768 * 12608 * 2);   // [3][b*h][197][64] bf16
    u16* xbf    = (u16*)carve(12608ull * 768 * 2);
    u16* qkvwb  = (u16*)carve(768ull * 2304 * 2);
    u16* projwb = (u16*)carve(768ull * 768 * 2);
    u16* transb = (u16*)carve(12ull * 196 * 196 * 2 + 256);
    u16* qprime  = xbf;      // alias: xbf dead after QKV GEMM
    u16* attnout = qkvbuf;   // alias: q-region dead after q-transform

    cvt_kernel<<<(12608 * 768 + 255) / 256, 256, 0, stream>>>(x, xbf, 12608 * 768);
    cvt_kernel<<<(768 * 2304 + 255) / 256, 256, 0, stream>>>(qkvw, qkvwb, 768 * 2304);
    cvt_kernel<<<(768 * 768 + 255) / 256, 256, 0, stream>>>(projw, projwb, 768 * 768);
    trans_kernel<<<(196 * 196 + 255) / 256, 256, 0, stream>>>(stab, spar, transb);

    gemm_bf16<0><<<dim3(18, 99, 1), 256, 0, stream>>>(xbf, qkvwb, qkvb, (void*)qkvbuf);
    copy_cls_kernel<<<(768 * 64 + 255) / 256, 256, 0, stream>>>(qkvbuf, qprime);
    gemm_bf16<2><<<dim3(1, 2, 768), 256, 0, stream>>>(transb, qkvbuf, nullptr, (void*)qprime);

    attn_kernel<<<768, 128, ATTN_SMEM, stream>>>(qprime, qkvbuf, attnout);

    gemm_bf16<1><<<dim3(6, 99, 1), 256, 0, stream>>>(attnout, projwb, projb, d_out);
}
